// HybridDetectionModel_19284403159215
// MI455X (gfx1250) — compile-verified, matrix operand order checked by reading, not run on hardware
//
#include <hip/hip_runtime.h>
#include <hip/hip_bf16.h>

#define N_ANCH   250000
#define NCLS     80
#define KSEL     2048
#define NBINS    4096
#define CONF_TH  0.05f

typedef __attribute__((ext_vector_type(16))) _Float16 v16h;
typedef __attribute__((ext_vector_type(8)))  float    v8f;

__device__ __forceinline__ float sigmoidf_(float x) { return 1.0f / (1.0f + __expf(-x)); }

// ---------------------------------------------------------------- zero control
__global__ __launch_bounds__(256) void zero_kernel(int* hist, int* ctl) {
    int i = blockIdx.x * 256 + threadIdx.x;
    if (i < NBINS) hist[i] = 0;
    if (i < 8) ctl[i] = 0;   // c0, c1, cutBin, cntAbove, maxcoord_bits, ...
}

// ------------------------------------------------- per-anchor max score + label
// one wave32 per anchor; lanes stripe the 80 classes (coalesced bursts).
__global__ __launch_bounds__(256) void score_kernel(const float* __restrict__ logits,
                                                    const float* __restrict__ ctr,
                                                    float* __restrict__ maxs,
                                                    int* __restrict__ lbls) {
    int wave = (blockIdx.x * 256 + threadIdx.x) >> 5;
    int lane = threadIdx.x & 31;
    if (wave >= N_ANCH) return;
    const float* row = logits + (size_t)wave * NCLS;
    float m = row[lane];
    int   l = lane;
    float v1 = row[lane + 32];
    if (v1 > m) { m = v1; l = lane + 32; }
    if (lane < 16) { float v2 = row[lane + 64]; if (v2 > m) { m = v2; l = lane + 64; } }
#pragma unroll
    for (int off = 16; off > 0; off >>= 1) {
        float om = __shfl_xor(m, off, 32);
        int   ol = __shfl_xor(l, off, 32);
        if (om > m || (om == m && ol < l)) { m = om; l = ol; }
    }
    if (lane == 0) {
        // sigmoid is monotone: argmax(logits)==argmax(scores)
        maxs[wave] = sigmoidf_(m) * sigmoidf_(ctr[wave]);
        lbls[wave] = l;
    }
}

// ------------------------------------------------------------------- histogram
__global__ __launch_bounds__(256) void hist_kernel(const float* __restrict__ maxs,
                                                   int* __restrict__ hist) {
    __shared__ int lh[NBINS];
    for (int i = threadIdx.x; i < NBINS; i += 256) lh[i] = 0;
    __syncthreads();
    for (int i = blockIdx.x * 256 + threadIdx.x; i < N_ANCH; i += gridDim.x * 256) {
        float s = maxs[i];
        if (s > CONF_TH) {
            int b = (int)(s * (float)NBINS);
            b = b < 0 ? 0 : (b > NBINS - 1 ? NBINS - 1 : b);
            atomicAdd(&lh[b], 1);
        }
    }
    __syncthreads();
    for (int i = threadIdx.x; i < NBINS; i += 256)
        if (lh[i]) atomicAdd(&hist[i], lh[i]);
}

// ------------------------------------------- find cutoff bin + init selection
__global__ __launch_bounds__(256) void scan_kernel(const int* __restrict__ hist,
                                                   int* __restrict__ ctl,
                                                   float* __restrict__ selScore,
                                                   int* __restrict__ selIdx) {
    for (int i = threadIdx.x; i < KSEL; i += 256) {
        selScore[i] = -__builtin_inff();
        selIdx[i]   = 0;
    }
    if (threadIdx.x == 0) {
        int cum = 0, cut = -1;
        for (int b = NBINS - 1; b >= 0; --b) {
            int h = hist[b];
            if (cum + h > KSEL) { cut = b; break; }
            cum += h;
        }
        ctl[2] = cut;   // cutoff bin (elements in bins > cut all selected)
        ctl[3] = cum;   // count strictly above cutoff bin
    }
}

// ----------------------------------------------------------------- compaction
__global__ __launch_bounds__(256) void compact_kernel(const float* __restrict__ maxs,
                                                      int* __restrict__ ctl,
                                                      float* __restrict__ selScore,
                                                      int* __restrict__ selIdx) {
    int cut   = ctl[2];
    int above = ctl[3];
    for (int i = blockIdx.x * 256 + threadIdx.x; i < N_ANCH; i += gridDim.x * 256) {
        float s = maxs[i];
        if (s > CONF_TH) {
            int b = (int)(s * (float)NBINS);
            b = b > NBINS - 1 ? NBINS - 1 : b;
            if (b > cut) {
                int p = atomicAdd(&ctl[0], 1);           // p < above <= KSEL
                selScore[p] = s; selIdx[p] = i;
            } else if (b == cut) {
                int p = atomicAdd(&ctl[1], 1);
                if (p < KSEL - above) { selScore[above + p] = s; selIdx[above + p] = i; }
            }
        }
    }
}

// ---------------------------------------------------- bitonic sort 2048 in LDS
__device__ __forceinline__ bool greaterKey(float a, int ai, float b, int bi) {
    return (a > b) || (a == b && ai < bi);
}
__global__ __launch_bounds__(1024) void sort_kernel(float* __restrict__ selScore,
                                                    int* __restrict__ selIdx) {
    __shared__ float ss[KSEL];
    __shared__ int   si[KSEL];
    int t = threadIdx.x;
    ss[t] = selScore[t];              si[t] = selIdx[t];
    ss[t + 1024] = selScore[t + 1024]; si[t + 1024] = selIdx[t + 1024];
    for (int kk = 2; kk <= KSEL; kk <<= 1) {
        for (int j = kk >> 1; j > 0; j >>= 1) {
            __syncthreads();
#pragma unroll
            for (int base = 0; base < 2; ++base) {
                int i   = t + base * 1024;
                int ixj = i ^ j;
                if (ixj > i) {
                    float a = ss[i], b = ss[ixj];
                    int  ai = si[i], bi = si[ixj];
                    bool dir = ((i & kk) == 0);               // descending region
                    bool sw  = dir ? greaterKey(b, bi, a, ai)
                                   : greaterKey(a, ai, b, bi);
                    if (sw) { ss[i] = b; ss[ixj] = a; si[i] = bi; si[ixj] = ai; }
                }
            }
        }
    }
    __syncthreads();
    selScore[t] = ss[t];              selIdx[t] = si[t];
    selScore[t + 1024] = ss[t + 1024]; selIdx[t + 1024] = si[t + 1024];
}

// -------------------------------------------- gather top-K, emit most outputs
__global__ __launch_bounds__(256) void gather_kernel(const float* __restrict__ logits,
                                                     const float* __restrict__ bbox,
                                                     const float* __restrict__ ctr,
                                                     const float* __restrict__ maxs,
                                                     const int* __restrict__ lbls,
                                                     const float* __restrict__ selScore,
                                                     const int* __restrict__ selIdx,
                                                     float* __restrict__ out,
                                                     float* __restrict__ bx1, float* __restrict__ by1,
                                                     float* __restrict__ bx2, float* __restrict__ by2,
                                                     int* __restrict__ labK, int* __restrict__ validArr,
                                                     int* __restrict__ ctl) {
    int k    = (blockIdx.x * 256 + threadIdx.x) >> 5;   // 0..2047
    int lane = threadIdx.x & 31;
    float s  = selScore[k];
    int  idx = selIdx[k];
    int valid = (s > -3.0e38f) ? 1 : 0;
    float c = sigmoidf_(ctr[idx]);
    const float* row = logits + (size_t)idx * NCLS;
    float* srow = out + 8192 + (size_t)k * NCLS;        // scores_k region
    srow[lane]      = c * sigmoidf_(row[lane]);
    srow[lane + 32] = c * sigmoidf_(row[lane + 32]);
    if (lane < 16) srow[lane + 64] = c * sigmoidf_(row[lane + 64]);
    if (lane == 0) {
        float x1 = bbox[(size_t)idx * 4 + 0], y1 = bbox[(size_t)idx * 4 + 1];
        float x2 = bbox[(size_t)idx * 4 + 2], y2 = bbox[(size_t)idx * 4 + 3];
        out[k * 4 + 0] = x1; out[k * 4 + 1] = y1;
        out[k * 4 + 2] = x2; out[k * 4 + 3] = y2;
        out[172032 + k] = maxs[idx];                    // ms_k
        int lab = lbls[idx];
        out[174080 + k] = (float)lab;                   // labels (as float)
        bx1[k] = x1; by1[k] = y1; bx2[k] = x2; by2[k] = y2;
        labK[k] = lab; validArr[k] = valid;
        float mc = valid ? fmaxf(fmaxf(x1, y1), fmaxf(x2, y2)) : 0.0f;
        atomicMax(&ctl[4], __float_as_int(mc));         // all coords >= 0
    }
}

// ------------------------------------------------ class-offset boxes and areas
// Writes packed float4 boxes followed (contiguously in ws) by the area array,
// forming one 10240-dword region consumed by the async LDS stage below.
__global__ __launch_bounds__(256) void offset_kernel(const float* __restrict__ bx1, const float* __restrict__ by1,
                                                     const float* __restrict__ bx2, const float* __restrict__ by2,
                                                     const int* __restrict__ labK, const int* __restrict__ ctl,
                                                     float4* __restrict__ obox,
                                                     float* __restrict__ areaArr) {
    int k = blockIdx.x * 256 + threadIdx.x;             // grid covers exactly KSEL
    float maxc = __int_as_float(ctl[4]);
    float off  = (float)labK[k] * (maxc + 1.0f);
    float a1 = bx1[k] + off, b1 = by1[k] + off;
    float a2 = bx2[k] + off, b2 = by2[k] + off;
    obox[k] = make_float4(a1, b1, a2, b2);
    areaArr[k] = (a2 - a1) * (b2 - b1);
}

// ------------- IoU suppression bitmask; union outer-sum via WMMA f16 tiles ---
// Stage all boxes+areas (40KB) into LDS with global_load_async_to_lds_b128,
// then one wave per 16-row x 32-col tile; two 16x16 WMMAs give area_i+area_j.
// IoU test is divide-free: inter > 0.5*max(union,1e-9).
__global__ __launch_bounds__(128) void mask_kernel(const float* __restrict__ gdata, // 10240 dwords
                                                   unsigned* __restrict__ mask) {
    __shared__ float sh[10240];                         // [0..8191] float4 boxes, [8192..] areas

    // ---- async copy: 128 threads x 20 x b128 = 40KB ----
    unsigned lbase = (unsigned)(unsigned long long)(void*)sh;   // low 32 bits = LDS offset
#pragma unroll
    for (int c = 0; c < 20; ++c) {
        int d = c * 512 + threadIdx.x * 4;              // dword index
        unsigned laddr = lbase + (unsigned)(d * 4);
        const float* g = gdata + d;
        asm volatile("global_load_async_to_lds_b128 %0, %1, off"
                     :: "v"(laddr), "v"(g) : "memory");
    }
    asm volatile("s_wait_asynccnt 0" ::: "memory");
    __syncthreads();

    const float4* shBox  = (const float4*)sh;
    const float*  shArea = sh + 8192;

    int wave = (blockIdx.x * 128 + threadIdx.x) >> 5;   // 0..8191
    int lane = threadIdx.x & 31;
    int rowTile = wave >> 6;                            // 0..127
    int colWord = wave & 63;                            // 0..63
    int i0 = rowTile << 4;
    int j0 = colWord << 5;
    int lm = lane & 15;
    bool lo = lane < 16;

    // A (16x32 f16): A[m][0]=area_row[m], A[m][1]=1. Lanes>=16 hold K=8..15,24..31 -> 0.
    float arow = shArea[i0 + lm];
    v16h A = {};
    A[0] = (_Float16)(lo ? arow : 0.0f);
    A[1] = (_Float16)(lo ? 1.0f : 0.0f);

    unsigned wlow[8]  = {0, 0, 0, 0, 0, 0, 0, 0};
    unsigned whigh[8] = {0, 0, 0, 0, 0, 0, 0, 0};
    int mbase = (lane >> 4) << 3;                       // 0 or 8

#pragma unroll
    for (int jt = 0; jt < 32; jt += 16) {
        // B (32x16 f16): B[0][n]=1, B[1][n]=area_col[n]. Lanes>=16 hold K=8..15,24..31 -> 0.
        float acol = shArea[j0 + jt + lm];
        v16h B = {};
        B[0] = (_Float16)(lo ? 1.0f : 0.0f);
        B[1] = (_Float16)(lo ? acol : 0.0f);
        v8f C = {};
        C = __builtin_amdgcn_wmma_f32_16x16x32_f16(false, A, false, B, (short)0, C, false, false);

        int n = j0 + jt + lm;                           // this lane's column
        float4 cb = shBox[n];
#pragma unroll
        for (int r = 0; r < 8; ++r) {
            int i = i0 + mbase + r;                     // this lane's row for C[r]
            float4 rb = shBox[i];
            float xx1 = fmaxf(rb.x, cb.x), yy1 = fmaxf(rb.y, cb.y);
            float xx2 = fminf(rb.z, cb.z), yy2 = fminf(rb.w, cb.w);
            float inter = fmaxf(xx2 - xx1, 0.0f) * fmaxf(yy2 - yy1, 0.0f);
            float uni   = C[r] - inter;                 // area_i + area_j - inter
            // iou > 0.5  <=>  inter > 0.5 * max(uni, 1e-9)   (divide-free)
            bool pred   = (inter > 0.5f * fmaxf(uni, 1e-9f)) && (n > i);
            unsigned bal = (unsigned)__ballot(pred);    // wave32: low 32 bits
            unsigned blo = bal & 0xFFFFu;               // rows m=r     (lanes 0-15)
            unsigned bhi = bal >> 16;                   // rows m=r+8   (lanes 16-31)
            if (jt == 0) { wlow[r]  = blo;         whigh[r]  = bhi; }
            else         { wlow[r] |= blo << 16;   whigh[r] |= bhi << 16; }
        }
    }
    if (lane == 0) {
#pragma unroll
        for (int r = 0; r < 8; ++r) {
            mask[(size_t)(i0 + r)     * 64 + colWord] = wlow[r];
            mask[(size_t)(i0 + 8 + r) * 64 + colWord] = whigh[r];
        }
    }
}

// ---------------------------------------------------- serial greedy NMS reduce
__global__ __launch_bounds__(64) void nms_kernel(const unsigned* __restrict__ mask,
                                                 const int* __restrict__ validArr,
                                                 float* __restrict__ keepOut) {
    __shared__ unsigned rem[64];
    int tid = threadIdx.x;
    unsigned w = 0;
#pragma unroll 4
    for (int b = 0; b < 32; ++b)
        if (!validArr[tid * 32 + b]) w |= (1u << b);
    rem[tid] = w;
    __syncthreads();
    for (int i = 0; i < KSEL; ++i) {
        bool alive = ((rem[i >> 5] >> (i & 31)) & 1u) == 0u;
        __syncthreads();
        if (alive) rem[tid] |= mask[(size_t)i * 64 + tid];   // mask has only j>i bits
        __syncthreads();
    }
    unsigned fw = rem[tid];
#pragma unroll 4
    for (int b = 0; b < 32; ++b)
        keepOut[tid * 32 + b] = ((fw >> b) & 1u) ? 0.0f : 1.0f;
}

// =============================================================================
extern "C" void kernel_launch(void* const* d_in, const int* in_sizes, int n_in,
                              void* d_out, int out_size, void* d_ws, size_t ws_size,
                              hipStream_t stream) {
    const float* logits = (const float*)d_in[0];   // (1, 250000, 80)
    const float* bbox   = (const float*)d_in[1];   // (1, 250000, 4)
    const float* ctr    = (const float*)d_in[2];   // (1, 250000)
    float* out = (float*)d_out;

    size_t off = 0;
    char* base = (char*)d_ws;
    auto alloc = [&](size_t bytes) -> void* {
        void* p = base + off;
        off += (bytes + 255) & ~(size_t)255;
        return p;
    };
    float*    maxs     = (float*)alloc((size_t)N_ANCH * 4);
    int*      lbls     = (int*)  alloc((size_t)N_ANCH * 4);
    int*      hist     = (int*)  alloc((size_t)NBINS * 4);
    int*      ctl      = (int*)  alloc(8 * 4);
    float*    selScore = (float*)alloc((size_t)KSEL * 4);
    int*      selIdx   = (int*)  alloc((size_t)KSEL * 4);
    float*    bx1      = (float*)alloc((size_t)KSEL * 4);
    float*    by1      = (float*)alloc((size_t)KSEL * 4);
    float*    bx2      = (float*)alloc((size_t)KSEL * 4);
    float*    by2      = (float*)alloc((size_t)KSEL * 4);
    int*      labK     = (int*)  alloc((size_t)KSEL * 4);
    int*      validA   = (int*)  alloc((size_t)KSEL * 4);
    // obox (32768 B, multiple of 256) is immediately followed by areaArr,
    // forming one contiguous 10240-dword region for the async LDS stage.
    float4*   obox     = (float4*)alloc((size_t)KSEL * 16);
    float*    areaArr  = (float*)alloc((size_t)KSEL * 4);
    unsigned* maskBuf  = (unsigned*)alloc((size_t)KSEL * 64 * 4);
    (void)ws_size; (void)in_sizes; (void)n_in; (void)out_size;

    zero_kernel<<<(NBINS + 255) / 256, 256, 0, stream>>>(hist, ctl);
    score_kernel<<<N_ANCH / 8, 256, 0, stream>>>(logits, ctr, maxs, lbls);
    hist_kernel<<<512, 256, 0, stream>>>(maxs, hist);
    scan_kernel<<<1, 256, 0, stream>>>(hist, ctl, selScore, selIdx);
    compact_kernel<<<512, 256, 0, stream>>>(maxs, ctl, selScore, selIdx);
    sort_kernel<<<1, 1024, 0, stream>>>(selScore, selIdx);
    gather_kernel<<<KSEL / 8, 256, 0, stream>>>(logits, bbox, ctr, maxs, lbls,
                                                selScore, selIdx, out,
                                                bx1, by1, bx2, by2, labK, validA, ctl);
    offset_kernel<<<KSEL / 256, 256, 0, stream>>>(bx1, by1, bx2, by2, labK, ctl,
                                                  obox, areaArr);
    mask_kernel<<<(128 * 64) / 4, 128, 0, stream>>>((const float*)obox, maskBuf);
    nms_kernel<<<1, 64, 0, stream>>>(maskBuf, validA, out + 176128);
}
